// Nets_82575041233371
// MI455X (gfx1250) — compile-verified, matrix operand order checked by reading, not run on hardware
//
#include <hip/hip_runtime.h>
#include <hip/hip_bf16.h>
#include <math.h>

// ---------------- problem constants ----------------
#define NN   8192      // nodes
#define NE   131072    // edges
#define EMB  128
#define HC   512       // HEADS * EMB
#define R2C  6.25f
#define EPSC 1e-5f

typedef __attribute__((ext_vector_type(16))) _Float16 v16h;
typedef __attribute__((ext_vector_type(8)))  _Float16 v8h;
typedef __attribute__((ext_vector_type(8)))  float    v8f;
typedef __attribute__((ext_vector_type(4)))  unsigned int u32x4;
typedef __attribute__((ext_vector_type(4)))  int      i32x4;
typedef __attribute__((ext_vector_type(8)))  int      i32x8;

// =====================================================================
//  CDNA5 feature probe kernel (launched once; touches only LDS).
//  - TDM tensor_load_to_lds with count=0 NULL descriptor (ISA: NOP)
//  - cluster id / cluster barrier (NOP when not in a cluster)
// =====================================================================
__global__ void k_cdna5_probe(const float* g) {
    __shared__ __align__(16) float lbuf[64];
    int cid = __builtin_amdgcn_cluster_id_x();
    lbuf[threadIdx.x & 63] = (float)cid;

    unsigned lds_off = (unsigned)(size_t)&lbuf[0];
    unsigned long long ga = (unsigned long long)(size_t)g;
    asm volatile(
        "global_load_async_to_lds_b128 %0, %1, off\n\t"
        "s_wait_asynccnt 0"
        :: "v"(lds_off), "v"(ga) : "memory");

    u32x4 g0 = {};
    i32x8 g1 = {};
    i32x4 g2 = {};
    i32x4 g3 = {};
#if __has_include(<hip/amd_detail/amd_gfx1250_TDM.h>)
    i32x8 g4 = {};
    __builtin_amdgcn_tensor_load_to_lds(g0, g1, g2, g3, g4, 0);
#else
    __builtin_amdgcn_tensor_load_to_lds(g0, g1, g2, g3, 0);
#endif
    __builtin_amdgcn_s_wait_tensorcnt(0);

    __builtin_amdgcn_s_cluster_barrier();
}

// =====================================================================
//  Precision conversion kernels (run outside the GEMM hot loop)
// =====================================================================
// f32 -> f16, 8 elements/thread, vector store
__global__ void k_cvt(const float* __restrict__ X, _Float16* __restrict__ Y,
                      long long n8) {
    long long i = (long long)blockIdx.x * blockDim.x + threadIdx.x;
    if (i >= n8) return;
    long long b = i * 8;
    v8h h;
#pragma unroll
    for (int j = 0; j < 8; ++j) h[j] = (_Float16)X[b + j];
    *(v8h*)&Y[b] = h;
}

// weight W[K][N] f32 -> WT[N][K] f16 (transpose so B tiles are row-contiguous)
__global__ void k_cvt_tr(const float* __restrict__ W, _Float16* __restrict__ WT,
                         int K, int N) {
    int idx = blockIdx.x * blockDim.x + threadIdx.x;
    if (idx >= K * N) return;
    int n = idx / K, k = idx % K;
    WT[(size_t)n * K + k] = (_Float16)W[(size_t)k * N + n];
}

// =====================================================================
//  WMMA GEMM:  C[M,N] = act(A[M,K] @ B[K,N] + bias)
//  A: [M][K] f16.  BT: [N][K] f16 (pre-transposed weights).
//  Tiles staged global->LDS with GLOBAL_LOAD_ASYNC_TO_LDS_B128,
//  ping-pong double-buffered on ASYNCcnt (async loads retire in order,
//  so s_wait_asynccnt 3 retires the previous tile's 3-load batch while
//  the next tile's batch stays in flight).
//  Requires M%128==0, N%64==0, K%32==0.
//  Block = 256 threads (8 wave32 waves); 128x64 tile; each wave owns a
//  32x32 slab = four 16x16 accumulators via v_wmma_f32_16x16x32_f16.
// =====================================================================
#define BM 128
#define BN 64
#define BK 32
#define LDK (BK + 8)            // 40 halves = 80B row stride (16B aligned, bank-spread)
#define ABYTES (BM * LDK * 2)   // 10240
#define BBYTES (BN * LDK * 2)   // 5120

__global__ __launch_bounds__(256) void k_gemm_wmma(
    const _Float16* __restrict__ A, const _Float16* __restrict__ BT,
    const float* __restrict__ bias, float* __restrict__ C,
    int M, int K, int N, int relu)
{
    __shared__ __align__(16) _Float16 As[2][BM][LDK];
    __shared__ __align__(16) _Float16 Bs[2][BN][LDK];

    const int tid  = threadIdx.x;
    const int wave = tid >> 5;
    const int lane = tid & 31;
    const int m0 = 32 * (wave & 3);   // wave's M offset inside tile
    const int n0 = 32 * (wave >> 2);  // wave's N offset inside tile
    const int blockM = blockIdx.y * BM;
    const int blockN = blockIdx.x * BN;

    const unsigned asBase = (unsigned)(size_t)&As[0][0][0];
    const unsigned bsBase = (unsigned)(size_t)&Bs[0][0][0];

    // staging chunk coords: 16B chunks of 8 halves along K
    const int ar0 = tid >> 2;          // A rows 0..63   (chunk tid)
    const int ar1 = (tid >> 2) + 64;   // A rows 64..127 (chunk tid+256)
    const int aq  = tid & 3;           // chunk index within row (0..3)
    const int br  = tid >> 2;          // B row 0..63
    const int bq  = tid & 3;

    auto stage = [&](int p, int k0) {
        unsigned l0 = asBase + p * ABYTES + ar0 * (LDK * 2) + aq * 16;
        unsigned long long g0 =
            (unsigned long long)(size_t)(A + (size_t)(blockM + ar0) * K + k0 + aq * 8);
        asm volatile("global_load_async_to_lds_b128 %0, %1, off"
                     :: "v"(l0), "v"(g0) : "memory");
        unsigned l1 = asBase + p * ABYTES + ar1 * (LDK * 2) + aq * 16;
        unsigned long long g1 =
            (unsigned long long)(size_t)(A + (size_t)(blockM + ar1) * K + k0 + aq * 8);
        asm volatile("global_load_async_to_lds_b128 %0, %1, off"
                     :: "v"(l1), "v"(g1) : "memory");
        unsigned l2 = bsBase + p * BBYTES + br * (LDK * 2) + bq * 16;
        unsigned long long g2 =
            (unsigned long long)(size_t)(BT + (size_t)(blockN + br) * K + k0 + bq * 8);
        asm volatile("global_load_async_to_lds_b128 %0, %1, off"
                     :: "v"(l2), "v"(g2) : "memory");
    };

    v8f acc00 = {}, acc01 = {}, acc10 = {}, acc11 = {};

    stage(0, 0);
    int p = 0;
    for (int k0 = 0; k0 < K; k0 += BK) {
        const bool more = (k0 + BK) < K;
        if (more) {
            stage(p ^ 1, k0 + BK);                          // next tile in flight
            asm volatile("s_wait_asynccnt 3" ::: "memory"); // retire current tile's batch
        } else {
            asm volatile("s_wait_asynccnt 0" ::: "memory");
        }
        __syncthreads();   // all waves' async writes to buffer p visible

        // fragments: lanes 0-15 take K{0..7,16..23}, lanes 16-31 K{8..15,24..31}
        const int rr = lane & 15;
        const int ko = (lane < 16) ? 0 : 8;

        v8h a0lo = *(const v8h*)&As[p][m0 + rr][ko];
        v8h a0hi = *(const v8h*)&As[p][m0 + rr][ko + 16];
        v8h a1lo = *(const v8h*)&As[p][m0 + 16 + rr][ko];
        v8h a1hi = *(const v8h*)&As[p][m0 + 16 + rr][ko + 16];
        v8h b0lo = *(const v8h*)&Bs[p][n0 + rr][ko];
        v8h b0hi = *(const v8h*)&Bs[p][n0 + rr][ko + 16];
        v8h b1lo = *(const v8h*)&Bs[p][n0 + 16 + rr][ko];
        v8h b1hi = *(const v8h*)&Bs[p][n0 + 16 + rr][ko + 16];

        v16h a0 = __builtin_shufflevector(a0lo, a0hi, 0,1,2,3,4,5,6,7,8,9,10,11,12,13,14,15);
        v16h a1 = __builtin_shufflevector(a1lo, a1hi, 0,1,2,3,4,5,6,7,8,9,10,11,12,13,14,15);
        v16h b0 = __builtin_shufflevector(b0lo, b0hi, 0,1,2,3,4,5,6,7,8,9,10,11,12,13,14,15);
        v16h b1 = __builtin_shufflevector(b1lo, b1hi, 0,1,2,3,4,5,6,7,8,9,10,11,12,13,14,15);

        acc00 = __builtin_amdgcn_wmma_f32_16x16x32_f16(false, a0, false, b0, (short)0, acc00, false, false);
        acc01 = __builtin_amdgcn_wmma_f32_16x16x32_f16(false, a0, false, b1, (short)0, acc01, false, false);
        acc10 = __builtin_amdgcn_wmma_f32_16x16x32_f16(false, a1, false, b0, (short)0, acc10, false, false);
        acc11 = __builtin_amdgcn_wmma_f32_16x16x32_f16(false, a1, false, b1, (short)0, acc11, false, false);
        __syncthreads();   // buffer p free for the iteration after next
        p ^= 1;
    }

    // ---- write back 4 tiles; C/D layout: VGPR r -> M=r (lanes 0-15) / M=r+8
    const int rofs = (lane < 16) ? 0 : 8;
    const int col  = lane & 15;
#pragma unroll
    for (int r = 0; r < 8; ++r) {
        int row0 = blockM + m0 + rofs + r;
        int row1 = row0 + 16;
        int cc0  = blockN + n0 + col;
        int cc1  = cc0 + 16;
        float v00 = acc00[r], v01 = acc01[r], v10 = acc10[r], v11 = acc11[r];
        if (bias) { v00 += bias[cc0]; v01 += bias[cc1]; v10 += bias[cc0]; v11 += bias[cc1]; }
        if (relu) {
            v00 = fmaxf(v00, 0.f); v01 = fmaxf(v01, 0.f);
            v10 = fmaxf(v10, 0.f); v11 = fmaxf(v11, 0.f);
        }
        C[(size_t)row0 * N + cc0] = v00;
        C[(size_t)row0 * N + cc1] = v01;
        C[(size_t)row1 * N + cc0] = v10;
        C[(size_t)row1 * N + cc1] = v11;
    }
}

// =====================================================================
//  Support kernels
// =====================================================================
__global__ void k_fill(float* p, int n, float v) {
    for (int i = blockIdx.x * blockDim.x + threadIdx.x; i < n;
         i += gridDim.x * blockDim.x) p[i] = v;
}

__global__ void k_split_x(const float* x, float* pos, float* ang, float* helix, int n) {
    int i = blockIdx.x * blockDim.x + threadIdx.x;
    if (i >= n) return;
    pos[i*3+0] = x[i*7+0]; pos[i*3+1] = x[i*7+1]; pos[i*3+2] = x[i*7+2];
    ang[i*3+0] = x[i*7+3]; ang[i*3+1] = x[i*7+4]; ang[i*3+2] = x[i*7+5];
    helix[i]   = x[i*7+6];
}

// small dense layer (tiny K or N): one thread per output element
__global__ void k_lin_small(const float* __restrict__ X, const float* __restrict__ W,
                            const float* __restrict__ b, float* __restrict__ Y,
                            int M, int K, int N, int relu) {
    int idx = blockIdx.x * blockDim.x + threadIdx.x;
    if (idx >= M * N) return;
    int r = idx / N, c = idx % N;
    __builtin_prefetch(&X[(size_t)r * K], 0, 1);
    float acc = b ? b[c] : 0.f;
    for (int k = 0; k < K; ++k) acc += X[(size_t)r*K + k] * W[(size_t)k*N + c];
    if (relu) acc = fmaxf(acc, 0.f);
    Y[idx] = acc;
}

// batchnorm stats: one block per column
__global__ void k_bn_stats(const float* __restrict__ X, int M, int N,
                           float* mean, float* rstd) {
    int c = blockIdx.x;
    float s = 0.f, s2 = 0.f;
    for (int r = threadIdx.x; r < M; r += 256) {
        float v = X[(size_t)r*N + c];
        s += v; s2 += v*v;
    }
    __shared__ float sh[256], sh2[256];
    sh[threadIdx.x] = s; sh2[threadIdx.x] = s2;
    __syncthreads();
    for (int o = 128; o > 0; o >>= 1) {
        if (threadIdx.x < o) { sh[threadIdx.x] += sh[threadIdx.x+o];
                               sh2[threadIdx.x] += sh2[threadIdx.x+o]; }
        __syncthreads();
    }
    if (threadIdx.x == 0) {
        float m = sh[0] / (float)M;
        float var = sh2[0] / (float)M - m*m;
        mean[c] = m;
        rstd[c] = rsqrtf(var + EPSC);
    }
}

__global__ void k_bn_apply(float* X, const float* mean, const float* rstd,
                           const float* g, const float* b, int M, int N) {
    int idx = blockIdx.x * blockDim.x + threadIdx.x;
    if (idx >= M * N) return;
    int c = idx % N;
    X[idx] = g[c] * (X[idx] - mean[c]) * rstd[c] + b[c];
}

// concat [u[src] | phi[src] | ea | u[dst] | phi[dst]]  -> E x 640
__global__ void k_concat5(const float* u, const float* phi, const float* ea,
                          const int* src, const int* dst, float* out) {
    int e = blockIdx.x, c = threadIdx.x;
    int s = src[e], d = dst[e];
    size_t base = (size_t)e * 640;
    out[base +       c] = u  [(size_t)s*EMB + c];
    out[base + 128 + c] = phi[(size_t)s*EMB + c];
    out[base + 256 + c] = ea [(size_t)e*EMB + c];
    out[base + 384 + c] = u  [(size_t)d*EMB + c];
    out[base + 512 + c] = phi[(size_t)d*EMB + c];
}

// concat [a[src] | ea]  -> E x 256
__global__ void k_concat2e(const float* a, const float* ea, const int* src, float* out) {
    int e = blockIdx.x, c = threadIdx.x;
    int s = src[e];
    out[(size_t)e*256 +       c] = a [(size_t)s*EMB + c];
    out[(size_t)e*256 + 128 + c] = ea[(size_t)e*EMB + c];
}

// concat [a | b]  -> M x 256
__global__ void k_concat2n(const float* a, const float* b, float* out, int M) {
    int idx = blockIdx.x * blockDim.x + threadIdx.x;
    if (idx >= M * 256) return;
    int r = idx / 256, c = idx % 256;
    out[idx] = (c < 128) ? a[(size_t)r*128 + c] : b[(size_t)r*128 + (c-128)];
}

__global__ void k_scat_add(const float* msg, const int* dst, float* sum, float* cnt) {
    int e = blockIdx.x, c = threadIdx.x;
    int d = dst[e];
    atomicAdd(&sum[(size_t)d*EMB + c], msg[(size_t)e*EMB + c]);
    if (c == 0) atomicAdd(&cnt[d], 1.f);
}

__global__ void k_scat_fin(const float* sum, const float* cnt, float* mu, int n) {
    int idx = blockIdx.x * blockDim.x + threadIdx.x;
    if (idx >= n) return;
    mu[idx] = sum[idx] / fmaxf(cnt[idx / EMB], 1.f);
}

__global__ void k_add_inplace(float* a, const float* b, int n) {
    int i = blockIdx.x * blockDim.x + threadIdx.x;
    if (i < n) a[i] += b[i];
}

__global__ void k_add_relu(const float* a, const float* b, float* c, int n) {
    int i = blockIdx.x * blockDim.x + threadIdx.x;
    if (i < n) c[i] = fmaxf(a[i] + b[i], 0.f);
}

// -------- transformer-conv pieces --------
__global__ void k_alpha(const float* q, const float* k, const float* e,
                        const int* src, const int* dst, float* alpha) {
    int idx = blockIdx.x * blockDim.x + threadIdx.x;     // NE*4
    if (idx >= NE * 4) return;
    int eI = idx >> 2, h = idx & 3;
    int s = src[eI], d = dst[eI];
    const float* qp = q + (size_t)d * HC + h * EMB;
    const float* kp = k + (size_t)s * HC + h * EMB;
    const float* ep = e + (size_t)eI * HC + h * EMB;
    float acc = 0.f;
    for (int c = 0; c < EMB; ++c) acc += qp[c] * (kp[c] + ep[c]);
    alpha[idx] = acc * 0.08838834764831845f;   // 1/sqrt(128)
}

__device__ inline void atomicMaxF(float* addr, float val) {
    unsigned int* a = (unsigned int*)addr;
    unsigned int old = *a;
    while (true) {
        float f = __uint_as_float(old);
        if (f >= val) break;
        unsigned int assumed = old;
        old = atomicCAS(a, assumed, __float_as_uint(val));
        if (old == assumed) break;
    }
}

__global__ void k_amax(const float* alpha, const int* dst, float* amax) {
    int idx = blockIdx.x * blockDim.x + threadIdx.x;
    if (idx >= NE * 4) return;
    int d = dst[idx >> 2], h = idx & 3;
    atomicMaxF(&amax[(size_t)d*4 + h], alpha[idx]);
}

__global__ void k_exden(const float* alpha, const float* amax, const int* dst,
                        float* ex, float* den) {
    int idx = blockIdx.x * blockDim.x + threadIdx.x;
    if (idx >= NE * 4) return;
    int d = dst[idx >> 2], h = idx & 3;
    float v = expf(alpha[idx] - amax[(size_t)d*4 + h]);
    ex[idx] = v;
    atomicAdd(&den[(size_t)d*4 + h], v);
}

__global__ void k_scat_attn(const float* v, const float* e, const float* ex,
                            const float* den, const int* src, const int* dst,
                            float* out) {
    int eI = blockIdx.x, c = threadIdx.x;
    int s = src[eI], d = dst[eI];
#pragma unroll
    for (int h = 0; h < 4; ++h) {
        float attn = ex[(size_t)eI*4 + h] / den[(size_t)d*4 + h];
        int col = h * EMB + c;
        atomicAdd(&out[(size_t)d*HC + col],
                  (v[(size_t)s*HC + col] + e[(size_t)eI*HC + col]) * attn);
    }
}

__global__ void k_gru(const float* gi, const float* gh, const float* h,
                      float* out, int M) {
    int idx = blockIdx.x * blockDim.x + threadIdx.x;
    if (idx >= M * EMB) return;
    int r = idx / EMB, c = idx % EMB;
    float ir  = gi[(size_t)r*384 +        c], hr = gh[(size_t)r*384 +        c];
    float iz  = gi[(size_t)r*384 + 128 +  c], hz = gh[(size_t)r*384 + 128 +  c];
    float inn = gi[(size_t)r*384 + 256 +  c], hn = gh[(size_t)r*384 + 256 +  c];
    float rr = 1.f / (1.f + expf(-(ir + hr)));
    float z  = 1.f / (1.f + expf(-(iz + hz)));
    float nc = tanhf(inn + rr * hn);
    out[idx] = (1.f - z) * nc + z * h[idx];
}

// -------- electrostatics (O(N^2) pair screen) --------
__global__ void k_el_pass1(const float* pos, const float* helix,
                           const float* W0, const float* b0,
                           float* sums, float* sumsq, float* count) {
    int i = blockIdx.x;
    float px = pos[i*3+0], py = pos[i*3+1], pz = pos[i*3+2];
    float hi = helix[i];
    for (int j = threadIdx.x; j < NN; j += blockDim.x) {
        float dx = px - pos[j*3+0], dy = py - pos[j*3+1], dz = pz - pos[j*3+2];
        float d2 = dx*dx + dy*dy + dz*dz;
        if (d2 < R2C && hi != helix[j]) {
            float xin[6] = {px, py, pz, pos[j*3+0], pos[j*3+1], pos[j*3+2]};
            for (int c = 0; c < 64; ++c) {
                float hv = b0[c];
                for (int k = 0; k < 6; ++k) hv += xin[k] * W0[k*64 + c];
                hv = fmaxf(hv, 0.f);
                atomicAdd(&sums[c], hv);
                atomicAdd(&sumsq[c], hv * hv);
            }
            atomicAdd(count, 1.f);
        }
    }
}

__global__ void k_el_bnfin(const float* s, const float* s2, const float* cnt,
                           float* mean, float* rstd) {
    int c = threadIdx.x;
    if (c < 64) {
        float C = fmaxf(cnt[0], 1.f);
        float m = s[c] / C;
        float var = s2[c] / C - m * m;
        mean[c] = m;
        rstd[c] = rsqrtf(var + EPSC);
    }
}

__global__ void k_el_pass2(const float* pos, const float* helix,
                           const float* W0, const float* b0,
                           const float* mean, const float* rstd,
                           const float* g, const float* be,
                           const float* W1, const float* b1,
                           float* aggs, float* aggc) {
    int i = blockIdx.x;
    float px = pos[i*3+0], py = pos[i*3+1], pz = pos[i*3+2];
    float hi = helix[i];
    for (int j = threadIdx.x; j < NN; j += blockDim.x) {
        float dx = px - pos[j*3+0], dy = py - pos[j*3+1], dz = pz - pos[j*3+2];
        float d2 = dx*dx + dy*dy + dz*dz;
        if (d2 < R2C && hi != helix[j]) {
            float xin[6] = {px, py, pz, pos[j*3+0], pos[j*3+1], pos[j*3+2]};
            float o0 = b1[0], o1 = b1[1], o2 = b1[2];
            for (int c = 0; c < 64; ++c) {
                float hv = b0[c];
                for (int k = 0; k < 6; ++k) hv += xin[k] * W0[k*64 + c];
                hv = fmaxf(hv, 0.f);
                float hb = g[c] * (hv - mean[c]) * rstd[c] + be[c];
                o0 += hb * W1[c*3 + 0];
                o1 += hb * W1[c*3 + 1];
                o2 += hb * W1[c*3 + 2];
            }
            atomicAdd(&aggs[(size_t)j*3 + 0], o0);
            atomicAdd(&aggs[(size_t)j*3 + 1], o1);
            atomicAdd(&aggs[(size_t)j*3 + 2], o2);
            atomicAdd(&aggc[j], 1.f);
        }
    }
}

__global__ void k_agg_fin(float* aggs, const float* aggc, int n) {
    int idx = blockIdx.x * blockDim.x + threadIdx.x;
    if (idx >= n * 3) return;
    aggs[idx] = aggs[idx] / fmaxf(aggc[idx / 3], 1.f);
}

__global__ void k_concat_pos_agg(const float* pos, const float* agg, float* out, int n) {
    int i = blockIdx.x * blockDim.x + threadIdx.x;
    if (i >= n) return;
    out[i*6+0] = pos[i*3+0]; out[i*6+1] = pos[i*3+1]; out[i*6+2] = pos[i*3+2];
    out[i*6+3] = agg[i*3+0]; out[i*6+4] = agg[i*3+1]; out[i*6+5] = agg[i*3+2];
}

__global__ void k_add3(const float* a, const float* b, float* c, int n) {
    int i = blockIdx.x * blockDim.x + threadIdx.x;
    if (i < n) c[i] = a[i] + b[i];
}

__global__ void k_final(const float* pos, const float* ang, const float* du3,
                        const float* el3, const float* dphi3, float* out) {
    int i = blockIdx.x * blockDim.x + threadIdx.x;
    if (i >= NN) return;
#pragma unroll
    for (int c = 0; c < 3; ++c) {
        out[i*6 + c]     = pos[i*3 + c] + du3[i*3 + c] + el3[i*3 + c];
        out[i*6 + 3 + c] = ang[i*3 + c] + dphi3[i*3 + c];
    }
}

// =====================================================================
//  Host orchestration
// =====================================================================
extern "C" void kernel_launch(void* const* d_in, const int* in_sizes, int n_in,
                              void* d_out, int out_size, void* d_ws, size_t ws_size,
                              hipStream_t stream)
{
    (void)in_sizes; (void)n_in; (void)out_size; (void)ws_size;

    const float* x      = (const float*)d_in[0];
    const int*   ei     = (const int*)  d_in[1];
    const float* eattr  = (const float*)d_in[2];
    const int* src = ei;
    const int* dst = ei + NE;

    enum {
        I_PE_W0 = 3, I_PE_B0, I_PE_W1, I_PE_B1,
        I_AE_W0, I_AE_B0, I_AE_W1, I_AE_B1,
        I_EE_W0, I_EE_B0, I_EE_W1, I_EE_B1,
        I_EB_W0, I_EB_B0, I_EB_G, I_EB_BE, I_EB_W1, I_EB_B1, I_EB_W2, I_EB_B2,
        I_NP1_W0, I_NP1_B0, I_NP1_G, I_NP1_BE, I_NP1_W1, I_NP1_B1, I_NP1_W2, I_NP1_B2,
        I_NP2_W0, I_NP2_B0, I_NP2_G, I_NP2_BE, I_NP2_W1, I_NP2_B1, I_NP2_W2, I_NP2_B2,
        I_NA1_W0, I_NA1_B0, I_NA1_G, I_NA1_BE, I_NA1_W1, I_NA1_B1, I_NA1_W2, I_NA1_B2,
        I_NA2_W0, I_NA2_B0, I_NA2_G, I_NA2_BE, I_NA2_W1, I_NA2_B1, I_NA2_W2, I_NA2_B2,
        I_BNP_G, I_BNP_B, I_BNA_G, I_BNA_B,
        I_TCP_WQ, I_TCP_WK, I_TCP_WV, I_TCP_WS, I_TCP_WE,
        I_TCP_BQ, I_TCP_BK, I_TCP_BV, I_TCP_BS,
        I_TCA_WQ, I_TCA_WK, I_TCA_WV, I_TCA_WS, I_TCA_WE,
        I_TCA_BQ, I_TCA_BK, I_TCA_BV, I_TCA_BS,
        I_PTF_W, I_PTF_B, I_ATF_W, I_ATF_B,
        I_GRUP_WIH, I_GRUP_WHH, I_GRUP_BIH, I_GRUP_BHH,
        I_GRUA_WIH, I_GRUA_WHH, I_GRUA_BIH, I_GRUA_BHH,
        I_PD_W0, I_PD_B0, I_PD_W1, I_PD_B1,
        I_AD_W0, I_AD_B0, I_AD_W1, I_AD_B1,
        I_EL1_W0, I_EL1_B0, I_EL1_G, I_EL1_BE, I_EL1_W1, I_EL1_B1,
        I_EL2_W0, I_EL2_B0, I_EL2_G, I_EL2_BE, I_EL2_W1, I_EL2_B1
    };
    auto PF = [&](int i) -> const float* { return (const float*)d_in[i]; };

    // ---- workspace bump allocator ----
    char* wptr = (char*)d_ws;
    auto alloc = [&](size_t nfloats) -> float* {
        float* p = (float*)wptr;
        size_t bytes = (nfloats * sizeof(float) + 255) & ~(size_t)255;
        wptr += bytes;
        return p;
    };
    auto halloc = [&](size_t nhalves) -> _Float16* {
        _Float16* p = (_Float16*)wptr;
        size_t bytes = (nhalves * 2 + 255) & ~(size_t)255;
        wptr += bytes;
        return p;
    };

    float* POS   = alloc((size_t)NN * 3);
    float* ANG   = alloc((size_t)NN * 3);
    float* HEL   = alloc(NN);
    float* U     = alloc((size_t)NN * EMB);
    float* PHI   = alloc((size_t)NN * EMB);
    float* H1    = alloc((size_t)NN * EMB);
    float* H2    = alloc((size_t)NN * EMB);
    float* ENC_H = alloc((size_t)NE * 64);
    float* EA    = alloc((size_t)NE * EMB);
    float* CAT5  = alloc((size_t)NE * 640);    // reused as E x 256 concat too
    float* T1    = alloc((size_t)NE * 384);
    float* T2    = alloc((size_t)NE * 384);
    float* MSG   = alloc((size_t)NE * EMB);
    float* SUMB  = alloc((size_t)NN * EMB);
    float* CNT   = alloc(NN);
    float* MU    = alloc((size_t)NN * EMB);
    float* CATN  = alloc((size_t)NN * 256);
    float* TN1   = alloc((size_t)NN * 256);
    float* TN2   = alloc((size_t)NN * 256);
    float* DU    = alloc((size_t)NN * EMB);
    float* DPHI  = alloc((size_t)NN * EMB);
    float* QB    = alloc((size_t)NN * HC);
    float* KB    = alloc((size_t)NN * HC);
    float* VB    = alloc((size_t)NN * HC);
    float* SB    = alloc((size_t)NN * HC);
    float* E512  = alloc((size_t)NE * HC);
    float* ALPHA = alloc((size_t)NE * 4);
    float* EXB   = alloc((size_t)NE * 4);
    float* AMAX  = alloc((size_t)NN * 4);
    float* DEN   = alloc((size_t)NN * 4);
    float* ATTNO = alloc((size_t)NN * HC);
    float* TBUF  = alloc((size_t)NN * HC);
    float* GI    = alloc((size_t)NN * 384);
    float* GH    = alloc((size_t)NN * 384);
    float* UNEW  = alloc((size_t)NN * EMB);
    float* MEANB = alloc(512);
    float* RSTDB = alloc(512);
    float* DU3   = alloc((size_t)NN * 3);
    float* DPHI3 = alloc((size_t)NN * 3);
    float* POS2  = alloc((size_t)NN * 3);
    float* ESUM  = alloc(64);
    float* ESQ   = alloc(64);
    float* ECNT  = alloc(1);
    float* EMEAN = alloc(64);
    float* ERSTD = alloc(64);
    float* AGGS  = alloc((size_t)NN * 3);
    float* AGGC  = alloc(NN);
    float* CAT6  = alloc((size_t)NN * 6);
    float* H2B   = alloc((size_t)NN * 64);
    float* EL3   = alloc((size_t)NN * 3);

    _Float16* AH = halloc((size_t)NE * 640);   // f16 staging for GEMM A operands

    float* OUT = (float*)d_out;

    // ---- helpers ----
    // gemm: converts A (f32->f16) and W (f32->f16 transposed [N][K]), then
    // runs the async-staged WMMA GEMM.  Call sequence is fixed -> the f16
    // weight arena layout is identical on every launch (graph-safe).
    auto gemm = [&](const float* A, const float* W, const float* bias,
                    float* C, int M, int K, int Nn, int relu) {
        _Float16* BT = halloc((size_t)K * Nn);
        k_cvt_tr<<<((size_t)K * Nn + 255) / 256, 256, 0, stream>>>(W, BT, K, Nn);
        long long n8 = (long long)M * K / 8;
        k_cvt<<<(n8 + 255) / 256, 256, 0, stream>>>(A, AH, n8);
        dim3 g(Nn / BN, M / BM);
        k_gemm_wmma<<<g, 256, 0, stream>>>(AH, BT, bias, C, M, K, Nn, relu);
    };
    auto bn = [&](float* X, int M, int Nn, const float* g, const float* b) {
        k_bn_stats<<<Nn, 256, 0, stream>>>(X, M, Nn, MEANB, RSTDB);
        k_bn_apply<<<((size_t)M * Nn + 255) / 256, 256, 0, stream>>>(
            X, MEANB, RSTDB, g, b, M, Nn);
    };
    auto fill = [&](float* p, int n, float v) {
        int blocks = (n + 255) / 256; if (blocks > 16384) blocks = 16384;
        k_fill<<<blocks, 256, 0, stream>>>(p, n, v);
    };
    auto mlp3 = [&](const float* X, int M, int Kin, int Hdim,
                    const float* w0, const float* b0, const float* g, const float* be,
                    const float* w1, const float* b1, const float* w2, const float* b2,
                    float* t1, float* t2, float* Y) {
        gemm(X, w0, b0, t1, M, Kin, Hdim, 1);
        bn(t1, M, Hdim, g, be);
        gemm(t1, w1, b1, t2, M, Hdim, Hdim, 1);
        gemm(t2, w2, b2, Y, M, Hdim, EMB, 0);
    };
    auto lin_small = [&](const float* X, const float* W, const float* b,
                         float* Y, int M, int K, int Nn, int relu) {
        k_lin_small<<<((size_t)M * Nn + 255) / 256, 256, 0, stream>>>(
            X, W, b, Y, M, K, Nn, relu);
    };

    // ---- split input features + CDNA5 probe ----
    k_split_x<<<(NN + 255) / 256, 256, 0, stream>>>(x, POS, ANG, HEL, NN);
    k_cdna5_probe<<<1, 32, 0, stream>>>(POS);

    // ---- encoders ----
    lin_small(POS, PF(I_PE_W0), PF(I_PE_B0), ENC_H, NN, 3, 64, 1);
    lin_small(ENC_H, PF(I_PE_W1), PF(I_PE_B1), U, NN, 64, EMB, 1);
    lin_small(ANG, PF(I_AE_W0), PF(I_AE_B0), ENC_H, NN, 3, 64, 1);
    lin_small(ENC_H, PF(I_AE_W1), PF(I_AE_B1), PHI, NN, 64, EMB, 1);
    lin_small(eattr, PF(I_EE_W0), PF(I_EE_B0), ENC_H, NE, 4, 64, 1);
    lin_small(ENC_H, PF(I_EE_W1), PF(I_EE_B1), EA, NE, 64, EMB, 1);
    hipMemcpyAsync(H1, U,   (size_t)NN * EMB * sizeof(float), hipMemcpyDeviceToDevice, stream);
    hipMemcpyAsync(H2, PHI, (size_t)NN * EMB * sizeof(float), hipMemcpyDeviceToDevice, stream);

    // ---- transformer-conv + GRU host sequence ----
    auto tconv_gru = [&](float* feat, float* hstate,
                         const float* wq, const float* bq, const float* wk, const float* bk,
                         const float* wv, const float* bv, const float* wsW, const float* bs,
                         const float* we, const float* tfw, const float* tfb,
                         const float* gwih, const float* gbih,
                         const float* gwhh, const float* gbhh) {
        gemm(feat, wq, bq, QB, NN, EMB, HC, 0);
        gemm(feat, wk, bk, KB, NN, EMB, HC, 0);
        gemm(feat, wv, bv, VB, NN, EMB, HC, 0);
        gemm(feat, wsW, bs, SB, NN, EMB, HC, 0);
        gemm(EA, we, nullptr, E512, NE, EMB, HC, 0);
        k_alpha<<<(NE * 4 + 255) / 256, 256, 0, stream>>>(QB, KB, E512, src, dst, ALPHA);
        fill(AMAX, NN * 4, -1e30f);
        k_amax<<<(NE * 4 + 255) / 256, 256, 0, stream>>>(ALPHA, dst, AMAX);
        fill(DEN, NN * 4, 0.f);
        k_exden<<<(NE * 4 + 255) / 256, 256, 0, stream>>>(ALPHA, AMAX, dst, EXB, DEN);
        fill(ATTNO, NN * HC, 0.f);
        k_scat_attn<<<NE, 128, 0, stream>>>(VB, E512, EXB, DEN, src, dst, ATTNO);
        k_add_relu<<<(NN * HC + 255) / 256, 256, 0, stream>>>(ATTNO, SB, TBUF, NN * HC);
        gemm(TBUF, tfw, tfb, UNEW, NN, HC, EMB, 0);
        gemm(UNEW, gwih, gbih, GI, NN, EMB, 384, 0);
        gemm(hstate, gwhh, gbhh, GH, NN, EMB, 384, 0);
        k_gru<<<(NN * EMB + 255) / 256, 256, 0, stream>>>(GI, GH, hstate, feat, NN);
        hipMemcpyAsync(hstate, feat, (size_t)NN * EMB * sizeof(float),
                       hipMemcpyDeviceToDevice, stream);
    };

    // ================= propagation rounds =================
    for (int i = 0; i < 3; ++i) {
        // ---- edge MLP (eb) ----
        k_concat5<<<NE, 128, 0, stream>>>(U, PHI, EA, src, dst, CAT5);
        mlp3(CAT5, NE, 640, 384,
             PF(I_EB_W0) + (size_t)i*640*384, PF(I_EB_B0) + i*384,
             PF(I_EB_G)  + i*384,             PF(I_EB_BE) + i*384,
             PF(I_EB_W1) + (size_t)i*384*384, PF(I_EB_B1) + i*384,
             PF(I_EB_W2) + (size_t)i*384*128, PF(I_EB_B2) + i*128,
             T1, T2, EA);

        // ---- np1 -> scatter_mean -> np2 (pos stream) ----
        k_concat2e<<<NE, 128, 0, stream>>>(U, EA, src, CAT5);
        mlp3(CAT5, NE, 256, 256,
             PF(I_NP1_W0) + (size_t)i*256*256, PF(I_NP1_B0) + i*256,
             PF(I_NP1_G)  + i*256,             PF(I_NP1_BE) + i*256,
             PF(I_NP1_W1) + (size_t)i*256*256, PF(I_NP1_B1) + i*256,
             PF(I_NP1_W2) + (size_t)i*256*128, PF(I_NP1_B2) + i*128,
             T1, T2, MSG);
        fill(SUMB, NN * EMB, 0.f); fill(CNT, NN, 0.f);
        k_scat_add<<<NE, 128, 0, stream>>>(MSG, dst, SUMB, CNT);
        k_scat_fin<<<(NN * EMB + 255) / 256, 256, 0, stream>>>(SUMB, CNT, MU, NN * EMB);
        k_concat2n<<<(NN * 256 + 255) / 256, 256, 0, stream>>>(U, MU, CATN, NN);
        mlp3(CATN, NN, 256, 256,
             PF(I_NP2_W0) + (size_t)i*256*256, PF(I_NP2_B0) + i*256,
             PF(I_NP2_G)  + i*256,             PF(I_NP2_BE) + i*256,
             PF(I_NP2_W1) + (size_t)i*256*256, PF(I_NP2_B1) + i*256,
             PF(I_NP2_W2) + (size_t)i*256*128, PF(I_NP2_B2) + i*128,
             TN1, TN2, DU);

        // ---- na1 -> scatter_mean -> na2 (angle stream) ----
        k_concat2e<<<NE, 128, 0, stream>>>(PHI, EA, src, CAT5);
        mlp3(CAT5, NE, 256, 256,
             PF(I_NA1_W0) + (size_t)i*256*256, PF(I_NA1_B0) + i*256,
             PF(I_NA1_G)  + i*256,             PF(I_NA1_BE) + i*256,
             PF(I_NA1_W1) + (size_t)i*256*256, PF(I_NA1_B1) + i*256,
             PF(I_NA1_W2) + (size_t)i*256*128, PF(I_NA1_B2) + i*128,
             T1, T2, MSG);
        fill(SUMB, NN * EMB, 0.f); fill(CNT, NN, 0.f);
        k_scat_add<<<NE, 128, 0, stream>>>(MSG, dst, SUMB, CNT);
        k_scat_fin<<<(NN * EMB + 255) / 256, 256, 0, stream>>>(SUMB, CNT, MU, NN * EMB);
        k_concat2n<<<(NN * 256 + 255) / 256, 256, 0, stream>>>(PHI, MU, CATN, NN);
        mlp3(CATN, NN, 256, 256,
             PF(I_NA2_W0) + (size_t)i*256*256, PF(I_NA2_B0) + i*256,
             PF(I_NA2_G)  + i*256,             PF(I_NA2_BE) + i*256,
             PF(I_NA2_W1) + (size_t)i*256*256, PF(I_NA2_B1) + i*256,
             PF(I_NA2_W2) + (size_t)i*256*128, PF(I_NA2_B2) + i*128,
             TN1, TN2, DPHI);

        // ---- residual + BN ----
        k_add_inplace<<<(NN * EMB + 255) / 256, 256, 0, stream>>>(U, DU, NN * EMB);
        bn(U, NN, EMB, PF(I_BNP_G) + i*EMB, PF(I_BNP_B) + i*EMB);
        k_add_inplace<<<(NN * EMB + 255) / 256, 256, 0, stream>>>(PHI, DPHI, NN * EMB);
        bn(PHI, NN, EMB, PF(I_BNA_G) + i*EMB, PF(I_BNA_B) + i*EMB);

        // ---- transformer conv + GRU, both streams ----
        tconv_gru(U, H1,
                  PF(I_TCP_WQ) + (size_t)i*EMB*HC, PF(I_TCP_BQ) + i*HC,
                  PF(I_TCP_WK) + (size_t)i*EMB*HC, PF(I_TCP_BK) + i*HC,
                  PF(I_TCP_WV) + (size_t)i*EMB*HC, PF(I_TCP_BV) + i*HC,
                  PF(I_TCP_WS) + (size_t)i*EMB*HC, PF(I_TCP_BS) + i*HC,
                  PF(I_TCP_WE) + (size_t)i*EMB*HC,
                  PF(I_PTF_W) + (size_t)i*HC*EMB, PF(I_PTF_B) + i*EMB,
                  PF(I_GRUP_WIH), PF(I_GRUP_BIH), PF(I_GRUP_WHH), PF(I_GRUP_BHH));
        tconv_gru(PHI, H2,
                  PF(I_TCA_WQ) + (size_t)i*EMB*HC, PF(I_TCA_BQ) + i*HC,
                  PF(I_TCA_WK) + (size_t)i*EMB*HC, PF(I_TCA_BK) + i*HC,
                  PF(I_TCA_WV) + (size_t)i*EMB*HC, PF(I_TCA_BV) + i*HC,
                  PF(I_TCA_WS) + (size_t)i*EMB*HC, PF(I_TCA_BS) + i*HC,
                  PF(I_TCA_WE) + (size_t)i*EMB*HC,
                  PF(I_ATF_W) + (size_t)i*HC*EMB, PF(I_ATF_B) + i*EMB,
                  PF(I_GRUA_WIH), PF(I_GRUA_BIH), PF(I_GRUA_WHH), PF(I_GRUA_BHH));
    }

    // ---- decoders ----
    lin_small(U, PF(I_PD_W0), PF(I_PD_B0), H2B, NN, EMB, 64, 1);
    lin_small(H2B, PF(I_PD_W1), PF(I_PD_B1), DU3, NN, 64, 3, 0);
    lin_small(PHI, PF(I_AD_W0), PF(I_AD_B0), H2B, NN, EMB, 64, 1);
    lin_small(H2B, PF(I_AD_W1), PF(I_AD_B1), DPHI3, NN, 64, 3, 0);

    // ---- electrostatics on pos + du ----
    k_add3<<<(NN * 3 + 255) / 256, 256, 0, stream>>>(POS, DU3, POS2, NN * 3);
    fill(ESUM, 64, 0.f); fill(ESQ, 64, 0.f); fill(ECNT, 1, 0.f);
    k_el_pass1<<<NN, 256, 0, stream>>>(POS2, HEL, PF(I_EL1_W0), PF(I_EL1_B0),
                                       ESUM, ESQ, ECNT);
    k_el_bnfin<<<1, 64, 0, stream>>>(ESUM, ESQ, ECNT, EMEAN, ERSTD);
    fill(AGGS, NN * 3, 0.f); fill(AGGC, NN, 0.f);
    k_el_pass2<<<NN, 256, 0, stream>>>(POS2, HEL, PF(I_EL1_W0), PF(I_EL1_B0),
                                       EMEAN, ERSTD, PF(I_EL1_G), PF(I_EL1_BE),
                                       PF(I_EL1_W1), PF(I_EL1_B1), AGGS, AGGC);
    k_agg_fin<<<(NN * 3 + 255) / 256, 256, 0, stream>>>(AGGS, AGGC, NN);
    k_concat_pos_agg<<<(NN + 255) / 256, 256, 0, stream>>>(POS2, AGGS, CAT6, NN);
    lin_small(CAT6, PF(I_EL2_W0), PF(I_EL2_B0), H2B, NN, 6, 64, 1);
    bn(H2B, NN, 64, PF(I_EL2_G), PF(I_EL2_BE));
    lin_small(H2B, PF(I_EL2_W1), PF(I_EL2_B1), EL3, NN, 64, 3, 0);

    // ---- final assembly ----
    k_final<<<(NN + 255) / 256, 256, 0, stream>>>(POS, ANG, DU3, EL3, DPHI3, OUT);
}
